// GlobalInformedSynchronyRouter_81243601371230
// MI455X (gfx1250) — compile-verified, matrix operand order checked by reading, not run on hardware
//
#include <hip/hip_runtime.h>
#include <hip/hip_bf16.h>

typedef __attribute__((ext_vector_type(16))) __bf16 v16bf;
typedef __attribute__((ext_vector_type(8)))  float  v8f;

#define EMBED 768
#define SPAT 128
#define TEMPD 64
#define SYNCD 32
#define GLOBD 64
#define NEXP 6
#define NLOC 224          // SPAT+TEMPD+SYNCD
#define NFEAT 288         // NLOC+GLOBD
#define KDIM 2304         // 3*EMBED
#define BATCH 8
#define SEQ 4096
#define ROWS (BATCH*SEQ)  // 32768
#define BPANEL (NLOC*32)  // bf16 elements per K-tile panel (7168) = 14336 bytes

// workspace layout (bytes)
#define WS_WBIG 0u                        // bf16, tiled [KDIM/32][NLOC][32]  ~1.0 MB
#define WS_YSUM (1u<<20)                  // f32 [BATCH][NLOC]
#define WS_GVEC (WS_YSUM + 8192u)         // f32 [BATCH][GLOBD]
#define WS_Y    (WS_GVEC + 4096u)         // f32 [ROWS][NLOC]    ~29.4 MB

// ---------------- K0: pack Wbig = [W_spat | W_temp | (0,+W_sync,-W_sync)] to bf16,
// pre-tiled into LDS-ready panels: dst[(k/32)*BPANEL + n*32 + (k%32)]; zero Ysum
__global__ __launch_bounds__(256) void pack_weights(
    const float* __restrict__ Wspat, const float* __restrict__ Wtemp,
    const float* __restrict__ Wsync, __bf16* __restrict__ WbigT, float* __restrict__ Ysum) {
  int idx = blockIdx.x * 256 + threadIdx.x;           // 0 .. KDIM*NLOC-1
  if (idx >= KDIM * NLOC) return;
  int k = idx / NLOC;
  int n = idx - k * NLOC;
  float v;
  if (n < SPAT) {
    v = Wspat[k * SPAT + n];
  } else if (n < SPAT + TEMPD) {
    v = Wtemp[k * TEMPD + (n - SPAT)];
  } else {
    int seg = k / EMBED, kr = k - seg * EMBED, c = n - (SPAT + TEMPD);
    v = (seg == 0) ? 0.0f : Wsync[kr * SYNCD + c];
    if (seg == 2) v = -v;
  }
  WbigT[(size_t)(k >> 5) * BPANEL + n * 32 + (k & 31)] = (__bf16)v;
  if (idx < BATCH * NLOC) Ysum[idx] = 0.0f;
}

// ---------------- K1: Y = bf16(Xcat) @ bf16(Wbig), f32 accum; fused column sums
__global__ __launch_bounds__(256) void gemm_wmma(
    const float* __restrict__ xt, const float* __restrict__ xa, const float* __restrict__ xv,
    const __bf16* __restrict__ WbigT, float* __restrict__ Y, float* __restrict__ Ysum) {
  __shared__ __bf16 sA[128 * 32];      // [m][k]  8 KB
  __shared__ __bf16 sB[NLOC * 32];     // [n][k]  14 KB
  const int tid  = threadIdx.x;
  const int wave = tid >> 5;
  const int lane = tid & 31;
  const int laneHalf = lane >> 4;      // 0: K lo halves, 1: K hi halves
  const int l16  = lane & 15;
  const int rowBase = blockIdx.x * 128;
  const unsigned sBoff = (unsigned)(uintptr_t)&sB[0];   // LDS byte offset (low 32 bits)

  v8f acc[14] = {};

  for (int kk = 0; kk < KDIM; kk += 32) {
    // --- B panel: async copy 14336 B straight into LDS (ASYNCcnt path) ---
    {
      const char* bsrc = (const char*)(WbigT + (size_t)(kk >> 5) * BPANEL);
      #pragma unroll
      for (int e = 0; e < 4; ++e) {
        int idx = tid + 256 * e;                    // 0..1023, need 0..895
        if (idx < (BPANEL * 2) / 16) {
          unsigned long long ga = (unsigned long long)(bsrc + idx * 16);
          unsigned la = sBoff + idx * 16;
          asm volatile("global_load_async_to_lds_b128 %0, %1, off"
                       :: "v"(la), "v"(ga) : "memory");
        }
      }
    }
    // --- A tile 128x32: float4 loads, bf16 pack, b64 LDS stores ---
    const int seg = kk / EMBED;
    const int ko  = kk - seg * EMBED;
    const float* src = (seg == 0) ? xt : (seg == 1) ? xa : xv;
    #pragma unroll
    for (int e = 0; e < 4; ++e) {
      int idx4 = tid + 256 * e;                     // 0..1023
      int m = idx4 >> 3, k4 = (idx4 & 7) << 2;
      const float4 f = *(const float4*)&src[(size_t)(rowBase + m) * EMBED + ko + k4];
      union { __bf16 h[4]; unsigned long long u64; } p;
      p.h[0] = (__bf16)f.x; p.h[1] = (__bf16)f.y;
      p.h[2] = (__bf16)f.z; p.h[3] = (__bf16)f.w;
      *(unsigned long long*)&sA[m * 32 + k4] = p.u64;
    }
    // prefetch next A tile (global_prefetch_b8)
    if (kk + 32 < KDIM) {
      int nk = kk + 32, nseg = nk / EMBED, nko = nk - nseg * EMBED;
      const float* nsrc = (nseg == 0) ? xt : (nseg == 1) ? xa : xv;
      __builtin_prefetch(&nsrc[(size_t)(rowBase + (tid >> 1)) * EMBED + nko + (tid & 1) * 16], 0, 3);
    }
    asm volatile("s_wait_asynccnt 0x0" ::: "memory");
    __syncthreads();

    // A fragment: lane<16 -> M=l16, K={0..7,16..23}; lane>=16 -> K={8..15,24..31}
    union { v16bf v; unsigned u[8]; } afr;
    const int mRow = wave * 16 + l16;
    #pragma unroll
    for (int q = 0; q < 8; ++q) {
      int kb = (q < 4) ? (laneHalf * 8 + 2 * q) : (16 + laneHalf * 8 + 2 * (q - 4));
      afr.u[q] = *(const unsigned*)&sA[mRow * 32 + kb];
    }
    #pragma unroll
    for (int j = 0; j < 14; ++j) {
      union { v16bf v; unsigned u[8]; } bfr;
      const int nCol = j * 16 + l16;
      #pragma unroll
      for (int q = 0; q < 8; ++q) {
        int kb = (q < 4) ? (laneHalf * 8 + 2 * q) : (16 + laneHalf * 8 + 2 * (q - 4));
        bfr.u[q] = *(const unsigned*)&sB[nCol * 32 + kb];
      }
      acc[j] = __builtin_amdgcn_wmma_f32_16x16x32_bf16(
          false, afr.v, false, bfr.v, (short)0, acc[j], false, false);
    }
    __syncthreads();
  }

  // writeback + fused per-batch column sums (blocks never straddle batches: 4096 % 128 == 0)
  const int b = rowBase >> 12;
  #pragma unroll
  for (int j = 0; j < 14; ++j) {
    float colsum = 0.0f;
    #pragma unroll
    for (int r = 0; r < 8; ++r) {
      int row = rowBase + wave * 16 + r + laneHalf * 8;   // C/D: vgpr r -> M=r (+8 for lanes 16-31)
      Y[(size_t)row * NLOC + j * 16 + l16] = acc[j][r];
      colsum += acc[j][r];
    }
    atomicAdd(&Ysum[b * NLOC + j * 16 + l16], colsum);
  }
}

// ---------------- K2: per-batch global summary + 224x64 projection
__global__ __launch_bounds__(256) void global_proj(
    const float* __restrict__ Ysum, const float* __restrict__ Y,
    const float* __restrict__ bspat, const float* __restrict__ btemp,
    const float* __restrict__ bsync, const float* __restrict__ Wglob,
    const float* __restrict__ bglob, float* __restrict__ gvec) {
  __shared__ float summary[NLOC];
  const int b = blockIdx.x, t = threadIdx.x;
  const float invS = 1.0f / (float)SEQ;
  if (t < NLOC) {
    float v;
    if (t < SPAT) {
      v = Ysum[b * NLOC + t] * invS + bspat[t];
    } else if (t < SPAT + TEMPD) {
      // telescoping: mean of deltas = (Y[S-1] - Y[0]) / S
      v = (Y[(size_t)(b * SEQ + SEQ - 1) * NLOC + t] - Y[(size_t)(b * SEQ) * NLOC + t]) * invS
          + btemp[t - SPAT];
    } else {
      v = Ysum[b * NLOC + t] * invS + bsync[t - SPAT - TEMPD];
    }
    summary[t] = v;
  }
  __syncthreads();
  if (t < GLOBD) {
    float a = bglob[t];
    #pragma unroll 4
    for (int f = 0; f < NLOC; ++f) a += summary[f] * Wglob[f * GLOBD + t];
    gvec[b * GLOBD + t] = a;
  }
}

// ---------------- K3: gate_logits = [local_vec, gvec] @ W_route + b_route
__global__ __launch_bounds__(256) void route(
    const float* __restrict__ Y, const float* __restrict__ gvec,
    const float* __restrict__ bspat, const float* __restrict__ btemp,
    const float* __restrict__ bsync, const float* __restrict__ Wroute,
    const float* __restrict__ broute, float* __restrict__ out) {
  __shared__ float sW[NFEAT * NEXP];   // 6912 B
  __shared__ float sBias[NLOC];
  const int tid = threadIdx.x;
  for (int i = tid; i < NFEAT * NEXP; i += 256) sW[i] = Wroute[i];
  for (int i = tid; i < NLOC; i += 256)
    sBias[i] = (i < SPAT) ? bspat[i]
             : (i < SPAT + TEMPD) ? btemp[i - SPAT] : bsync[i - SPAT - TEMPD];
  __syncthreads();

  const int r = blockIdx.x * 256 + tid;
  const int b = r >> 12;
  const int s = r & (SEQ - 1);
  const float* yr = Y + (size_t)r * NLOC;
  const float* yp = yr - NLOC;    // only read when s > 0

  float acc[NEXP];
  #pragma unroll
  for (int o = 0; o < NEXP; ++o) acc[o] = broute[o];

  for (int f = 0; f < SPAT; ++f) {
    float feat = yr[f] + sBias[f];
    #pragma unroll
    for (int o = 0; o < NEXP; ++o) acc[o] += feat * sW[f * NEXP + o];
  }
  for (int f = SPAT; f < SPAT + TEMPD; ++f) {
    float feat = (s ? (yr[f] - yp[f]) : 0.0f) + sBias[f];
    #pragma unroll
    for (int o = 0; o < NEXP; ++o) acc[o] += feat * sW[f * NEXP + o];
  }
  for (int f = SPAT + TEMPD; f < NLOC; ++f) {
    float feat = yr[f] + sBias[f];
    #pragma unroll
    for (int o = 0; o < NEXP; ++o) acc[o] += feat * sW[f * NEXP + o];
  }
  const float* gv = gvec + b * GLOBD;
  for (int f = 0; f < GLOBD; ++f) {
    float feat = gv[f];
    #pragma unroll
    for (int o = 0; o < NEXP; ++o) acc[o] += feat * sW[(NLOC + f) * NEXP + o];
  }
  #pragma unroll
  for (int o = 0; o < NEXP; ++o) out[(size_t)r * NEXP + o] = acc[o];
}

extern "C" void kernel_launch(void* const* d_in, const int* in_sizes, int n_in,
                              void* d_out, int out_size, void* d_ws, size_t ws_size,
                              hipStream_t stream) {
  (void)in_sizes; (void)n_in; (void)out_size; (void)ws_size;
  const float* xt    = (const float*)d_in[0];
  const float* xa    = (const float*)d_in[1];
  const float* xv    = (const float*)d_in[2];
  const float* Wspat = (const float*)d_in[3];
  const float* bspat = (const float*)d_in[4];
  const float* Wtemp = (const float*)d_in[5];
  const float* btemp = (const float*)d_in[6];
  const float* Wsync = (const float*)d_in[7];
  const float* bsync = (const float*)d_in[8];
  const float* Wglob = (const float*)d_in[9];
  const float* bglob = (const float*)d_in[10];
  const float* Wroute= (const float*)d_in[11];
  const float* broute= (const float*)d_in[12];

  char* ws = (char*)d_ws;
  __bf16* WbigT = (__bf16*)(ws + WS_WBIG);
  float*  Ysum  = (float*)(ws + WS_YSUM);
  float*  gvec  = (float*)(ws + WS_GVEC);
  float*  Y     = (float*)(ws + WS_Y);
  float*  out   = (float*)d_out;

  pack_weights<<<(KDIM * NLOC + 255) / 256, 256, 0, stream>>>(Wspat, Wtemp, Wsync, WbigT, Ysum);
  gemm_wmma<<<ROWS / 128, 256, 0, stream>>>(xt, xa, xv, WbigT, Y, Ysum);
  global_proj<<<BATCH, 256, 0, stream>>>(Ysum, Y, bspat, btemp, bsync, Wglob, bglob, gvec);
  route<<<ROWS / 256, 256, 0, stream>>>(Y, gvec, bspat, btemp, bsync, Wroute, broute, out);
}
